// BahdanauAttention_69106023793233
// MI455X (gfx1250) — matrix operand order repaired and checked by reading, not compiled, not run on hardware
//
#include <hip/hip_runtime.h>
#include <hip/hip_bf16.h>

typedef __bf16 v16bf __attribute__((ext_vector_type(16)));
typedef float  v8f   __attribute__((ext_vector_type(8)));

#define B_  32
#define S_  4096
#define D_  512
#define U_  256

// round-to-nearest-even f32 -> bf16 bits
__device__ __forceinline__ unsigned short f2bf(float f) {
  union { float f; unsigned u; } c; c.f = f;
  unsigned r = c.u + 0x7FFFu + ((c.u >> 16) & 1u);
  return (unsigned short)(r >> 16);
}
__device__ __forceinline__ unsigned pk2(float a, float b) {
  return (unsigned)f2bf(a) | ((unsigned)f2bf(b) << 16);
}
__device__ __forceinline__ __bf16 bfbits(unsigned short s) {
  return __builtin_bit_cast(__bf16, s);
}

// ---------------------------------------------------------------------------
// Kernel A: w1q[b,u] = sum_k query[b,k] * W1[k,u]     (tiny, f32 scalar)
// ---------------------------------------------------------------------------
__global__ void k_w1q(const float* __restrict__ q, const float* __restrict__ W1,
                      float* __restrict__ w1q) {
  const int b = blockIdx.x, u = threadIdx.x;
  float acc = 0.f;
  for (int k = 0; k < D_; ++k) acc += q[b * D_ + k] * W1[k * U_ + u];
  w1q[b * U_ + u] = acc;
}

// ---------------------------------------------------------------------------
// Kernel B: fused  scores = tanh(value@W2 + w1q) @ Wo + bo
// grid = 2048 (64 rows each), block = 256 (8 waves), dyn LDS = 80KB
//   wave w -> mtile = w&3 (16 rows), N-half = w>>2 (8 N-tiles)
//   => only 8 v8f accumulators per wave (64 VGPRs): no spills.
// LDS A-frags: [mtile(4)][kc(4)][lane(32)][16 bf16]  = 16KB
// LDS B-frags: [kc(4)][ntile(16)][lane(32)][16 bf16] = 64KB
// A-frags here (16x32): lane L, h=L/16: m=L%16 ; e -> K=h*16+e
// B-frags here (32x16), same k order as A: lane L: n=L%16 ; e -> K=(L/16)*16+e
// C layout: lane L, VGPR v -> M = v + 8*(L/16), N = L%16
// ---------------------------------------------------------------------------
__global__ void k_scores(const float* __restrict__ value,
                         const float* __restrict__ W2,
                         const float* __restrict__ w1q,
                         const float* __restrict__ Wo,
                         const float* __restrict__ bo,
                         float* __restrict__ scores) {
  extern __shared__ __attribute__((aligned(32))) char smem[];
  char* ldsA = smem;               // 16KB
  char* ldsB = smem + 16 * 1024;   // 64KB

  const int tid = threadIdx.x;
  const int w   = tid >> 5;        // wave id
  const int L   = tid & 31;        // lane
  const int mt  = w & 3;           // M-tile owned by this wave
  const int nh  = w >> 2;          // N-half owned by this wave (0 or 1)
  const int r0  = blockIdx.x * 64;     // first flat row (b*S + s)
  const int b   = r0 >> 12;            // uniform: 4096 % 64 == 0

  v8f acc[8];
#pragma unroll
  for (int j = 0; j < 8; ++j) acc[j] = (v8f){0.f,0.f,0.f,0.f,0.f,0.f,0.f,0.f};

  for (int p = 0; p < 4; ++p) {
    // ---- stage value panel [64 rows x 128 cols] f32 -> bf16 A-frags ----
    {
      const float* src = value + (size_t)r0 * D_ + p * 128;
#pragma unroll 4
      for (int i = 0; i < 8; ++i) {
        const int qd = tid + 256 * i;      // float4 task 0..2047
        const int m  = qd >> 5;            // row in panel (0..63)
        const int kk = (qd & 31) << 2;     // col in panel (mult of 4)
        const float4 v4 = *(const float4*)(src + (size_t)m * D_ + kk);
        const int kc   = kk >> 5;
        const int kk32 = kk & 31;
        const int lane = (m & 15) | ((kk32 >> 4) << 4);
        const int e0   = kk32 & 15;
        uint2 pkd; pkd.x = pk2(v4.x, v4.y); pkd.y = pk2(v4.z, v4.w);
        *(uint2*)(ldsA + ((((m >> 4) << 2) + kc) << 10) + (lane << 5) + (e0 << 1)) = pkd;
      }
      // ---- stage W2 panel [128 x 256] f32 -> bf16 B-frags (k-pairs) ----
      const float* w2s = W2 + (size_t)p * 128 * U_;
#pragma unroll 4
      for (int i = 0; i < 64; ++i) {
        const int qd = tid + 256 * i;      // task 0..16383
        const int j  = qd >> 8;            // k-pair 0..63
        const int n  = qd & 255;
        const float a0 = w2s[(size_t)(2 * j) * U_ + n];
        const float a1 = w2s[(size_t)(2 * j + 1) * U_ + n];
        const int kk32 = (2 * j) & 31;
        const int kc   = (2 * j) >> 5;
        const int lane = (n & 15) | ((kk32 >> 4) << 4);
        const int e0   = kk32 & 15;
        const int nt   = n >> 4;
        *(unsigned*)(ldsB + (((kc << 4) + nt) << 10) + (lane << 5) + (e0 << 1)) =
            pk2(a0, a1);
      }
      if (p < 3) {  // prefetch next value panel (global_prefetch_b8)
        const float* nxt = value + (size_t)r0 * D_ + (p + 1) * 128;
        __builtin_prefetch(nxt + (size_t)(tid >> 2) * D_ + (tid & 3) * 32, 0, 1);
      }
    }
    __syncthreads();
    // ---- compute: A-frag hoisted per K-chunk, sweep this wave's 8 N-tiles ----
#pragma unroll
    for (int kc = 0; kc < 4; ++kc) {
      const v16bf a = *(const v16bf*)(ldsA + (((mt << 2) + kc) << 10) + (L << 5));
#pragma unroll
      for (int j = 0; j < 8; ++j) {
        const int nt = (nh << 3) + j;
        const v16bf bb = *(const v16bf*)(ldsB + (((kc << 4) + nt) << 10) + (L << 5));
        acc[j] = __builtin_amdgcn_wmma_f32_16x16x32_bf16(
            false, a, false, bb, (short)0, acc[j], false, false);
      }
    }
    __syncthreads();
  }

  // ---- epilogue: tanh(acc + w1q) * Wo, reduce over this wave's N columns ----
  const int nl = L & 15;
  const int half = L >> 4;  // lanes 0..15 -> M=v ; lanes 16..31 -> M=v+8
  float sacc[8] = {0.f, 0.f, 0.f, 0.f, 0.f, 0.f, 0.f, 0.f};
#pragma unroll
  for (int j = 0; j < 8; ++j) {
    const int u    = ((nh << 3) + j) * 16 + nl;
    const float wq = w1q[b * U_ + u];
    const float wo = Wo[u];
#pragma unroll
    for (int v = 0; v < 8; ++v)
      sacc[v] += tanhf(acc[j][v] + wq) * wo;
  }
#pragma unroll
  for (int v = 0; v < 8; ++v) {
    float x = sacc[v];
    x += __shfl_xor(x, 1);
    x += __shfl_xor(x, 2);
    x += __shfl_xor(x, 4);
    x += __shfl_xor(x, 8);   // sums stay within each 16-lane half
    sacc[v] = x;
  }
  // combine the two N-halves (waves w and w+4) through LDS
  __syncthreads();                      // all waves done reading frag LDS
  float* red = (float*)ldsA;            // 8 waves * 16 rows = 512B, reuse ldsA
  if (nl == 0) {
#pragma unroll
    for (int v = 0; v < 8; ++v) red[w * 16 + half * 8 + v] = sacc[v];
  }
  __syncthreads();
  if (w < 4 && nl == 0) {
    const float bias = bo[0];
#pragma unroll
    for (int v = 0; v < 8; ++v) {
      const int idx = half * 8 + v;
      scores[r0 + w * 16 + idx] = red[w * 16 + idx] + red[(w + 4) * 16 + idx] + bias;
    }
  }
}

// ---------------------------------------------------------------------------
// Kernel C: softmax over S per batch row. grid=32, block=256
// ---------------------------------------------------------------------------
__global__ void k_softmax(const float* __restrict__ scores, float* __restrict__ attn) {
  __shared__ float red[8];
  const int b = blockIdx.x, tid = threadIdx.x;
  const float* row = scores + (size_t)b * S_;
  float x[16];
  float m = -3.4e38f;
#pragma unroll
  for (int j = 0; j < 16; ++j) { x[j] = row[tid + 256 * j]; m = fmaxf(m, x[j]); }
  for (int msk = 1; msk < 32; msk <<= 1) m = fmaxf(m, __shfl_xor(m, msk));
  if ((tid & 31) == 0) red[tid >> 5] = m;
  __syncthreads();
  float bm = red[0];
#pragma unroll
  for (int i = 1; i < 8; ++i) bm = fmaxf(bm, red[i]);
  __syncthreads();
  float s = 0.f;
#pragma unroll
  for (int j = 0; j < 16; ++j) { x[j] = __expf(x[j] - bm); s += x[j]; }
  for (int msk = 1; msk < 32; msk <<= 1) s += __shfl_xor(s, msk);
  if ((tid & 31) == 0) red[tid >> 5] = s;
  __syncthreads();
  float bs = 0.f;
#pragma unroll
  for (int i = 0; i < 8; ++i) bs += red[i];
  const float inv = 1.f / bs;
#pragma unroll
  for (int j = 0; j < 16; ++j) attn[(size_t)b * S_ + tid + 256 * j] = x[j] * inv;
}

// ---------------------------------------------------------------------------
// Kernel D: attention_weights[b,d,c] = sum_s value[b,s,d] * attn[c,s]
// Computed as out_bT[c,d]: M=c(32 -> 2 M-tiles), N=d, K=s via bf16 WMMA.
// grid = B*32 N-tiles = 1024 blocks x 1 wave (32 threads).
// ---------------------------------------------------------------------------
__global__ void k_attw(const float* __restrict__ value, const float* __restrict__ attn,
                       float* __restrict__ out) {
  const int blk = blockIdx.x;
  const int b = blk >> 5, nt = blk & 31;
  const int L = threadIdx.x;
  const int nl = L & 15, half = L >> 4;
  v8f c0 = (v8f){0.f,0.f,0.f,0.f,0.f,0.f,0.f,0.f};
  v8f c1 = c0;
  const float* vbase = value + (size_t)b * S_ * D_ + nt * 16 + nl;

  for (int kt = 0; kt < 128; ++kt) {
    // B-frag: lane reads a 16-deep column of value[:, d] (lanes form 64B rows)
    v16bf bf;
    const float* vp = vbase + (size_t)(kt * 32 + half * 8) * D_;
#pragma unroll
    for (int j = 0; j < 16; ++j) bf[j] = bfbits(f2bf(vp[(size_t)(j + (j & 8)) * D_]));
    // A-frags: contiguous attn chunks (two M-tiles: c = nl, c = 16+nl)
    const float* a0p = attn + (size_t)nl * S_ + kt * 32 + half * 8;
    const float* a1p = a0p + (size_t)16 * S_;
    v16bf a0, a1;
#pragma unroll
    for (int j = 0; j < 8; ++j) {
      a0[j]     = bfbits(f2bf(a0p[j]));
      a0[j + 8] = bfbits(f2bf(a0p[j + 16]));
      a1[j]     = bfbits(f2bf(a1p[j]));
      a1[j + 8] = bfbits(f2bf(a1p[j + 16]));
    }
    c0 = __builtin_amdgcn_wmma_f32_16x16x32_bf16(false, a0, false, bf, (short)0, c0, false, false);
    c1 = __builtin_amdgcn_wmma_f32_16x16x32_bf16(false, a1, false, bf, (short)0, c1, false, false);
  }
  // D layout: lane L, VGPR v -> c = v + 8*half (+16 for c1), d = nt*16 + nl
  const int d = nt * 16 + nl;
  float* o = out + ((size_t)b * D_ + d) * B_;
#pragma unroll
  for (int v = 0; v < 8; ++v) {
    o[v + 8 * half]      = c0[v];
    o[16 + v + 8 * half] = c1[v];
  }
}

// ---------------------------------------------------------------------------
extern "C" void kernel_launch(void* const* d_in, const int* in_sizes, int n_in,
                              void* d_out, int out_size, void* d_ws, size_t ws_size,
                              hipStream_t stream) {
  const float* query = (const float*)d_in[0];
  const float* value = (const float*)d_in[1];
  // d_in[2] = target_mask (unused by the reference math)
  const float* W1 = (const float*)d_in[3];
  const float* W2 = (const float*)d_in[4];
  const float* Wo = (const float*)d_in[5];
  const float* bo = (const float*)d_in[6];

  float* out_scores = (float*)d_out;                 // [B,S] = 131072 floats
  float* out_attw   = out_scores + B_ * S_;          // [B,D,B] = 524288 floats

  float* ws_w1q  = (float*)d_ws;                     // [B,U]  32KB
  float* ws_attn = ws_w1q + B_ * U_;                 // [B,S] 512KB

  k_w1q<<<B_, U_, 0, stream>>>(query, W1, ws_w1q);
  k_scores<<<(B_ * S_) / 64, 256, 80 * 1024, stream>>>(value, W2, ws_w1q, Wo, bo,
                                                       out_scores);
  k_softmax<<<B_, 256, 0, stream>>>(out_scores, ws_attn);
  k_attw<<<B_ * 32, 32, 0, stream>>>(value, ws_attn, out_attw);
}
